// SRNN_84284438217021
// MI455X (gfx1250) — matrix operand order repaired and checked by reading, not compiled, not run on hardware
//
#include <hip/hip_runtime.h>
#include <stdint.h>

// ---------------------------------------------------------------------------
// MI455X (gfx1250) spatial RNN: X-axis tanh RNN then Y-axis tanh RNN.
// B=16, C=64, H=128, W=128, HX=HY=64.
//
// Design: the only serial dependence is the 128-step recurrence, run by
// latency-bound workgroups. Everything parallel (input projection + bias) is
// hoisted into a wide WMMA "pregemm" kernel that writes its D fragments in
// C-operand layout, so each serial step is just:
//   v8f pre load (reg double-buffered) -> 2x ds_load_b128 -> 2x v_wmma
//   (independent accumulators) -> v_tanh_f32 -> ds_store_b16 -> barrier.
// ---------------------------------------------------------------------------

typedef __attribute__((ext_vector_type(16))) _Float16 v16h;
typedef __attribute__((ext_vector_type(8)))  _Float16 v8h;
typedef __attribute__((ext_vector_type(8)))  float    v8f;

#define HSTR 72   // padded LDS row stride (halfs): 144B keeps 16B alignment,
                  // spreads banks across rows.

static __device__ __forceinline__ float dev_tanh(float x) {
#if defined(__has_builtin)
#if __has_builtin(__builtin_amdgcn_tanhf)
  return __builtin_amdgcn_tanhf(x);   // gfx1250 v_tanh_f32
#else
  return tanhf(x);
#endif
#else
  return tanhf(x);
#endif
}

static __device__ __forceinline__ v16h combine(v8h lo, v8h hi) {
  return __builtin_shufflevector(lo, hi, 0, 1, 2, 3, 4, 5, 6, 7, 8, 9, 10, 11,
                                 12, 13, 14, 15);
}

// A-fragment (16x32 f16) from an LDS [16][HSTR] tile.
// lanes 0-15 : K kbase+0..15
// lanes 16-31: K kbase+16..31
static __device__ __forceinline__ v16h load_a_lds(const _Float16* base, int m,
                                                  int ksel, int kbase) {
  return *(const v16h*)(base + m * HSTR + kbase + ksel * 16);
}

static __device__ __forceinline__ v8f wmma_acc(v16h a, v16h b, v8f c) {
  return __builtin_amdgcn_wmma_f32_16x16x32_f16(false, a, false, b, (short)0, c,
                                                false, false);
}

// ---------------------------------------------------------------------------
// Kernel 0: weights f32 -> f16 (row-major [n][k] == B[k][n] read along k),
// fold biases.
// wbuf: [0)Wih_x [4096)Whh_x [8192)Wih_y [12288)Whh_y ; bbuf: [0)bx [64)by
// ---------------------------------------------------------------------------
__global__ __launch_bounds__(256) void prep_weights_kernel(
    const float* __restrict__ Wih_x, const float* __restrict__ Whh_x,
    const float* __restrict__ bih_x, const float* __restrict__ bhh_x,
    const float* __restrict__ Wih_y, const float* __restrict__ Whh_y,
    const float* __restrict__ bih_y, const float* __restrict__ bhh_y,
    _Float16* __restrict__ wbuf, float* __restrict__ bbuf) {
  const int tid = threadIdx.x;
  for (int i = tid; i < 4096; i += 256) {
    wbuf[i]         = (_Float16)Wih_x[i];
    wbuf[4096 + i]  = (_Float16)Whh_x[i];
    wbuf[8192 + i]  = (_Float16)Wih_y[i];
    wbuf[12288 + i] = (_Float16)Whh_y[i];
  }
  if (tid < 64) {
    bbuf[tid]      = bih_x[tid] + bhh_x[tid];
    bbuf[64 + tid] = bih_y[tid] + bhh_y[tid];
  }
}

// ---------------------------------------------------------------------------
// Kernel 1: x [B,C,H,W] f32 -> xh [B,H,W,C] f16 (LDS-tiled transpose).
// One block per (b,h): transpose the [C=64][W=128] slab.
// ---------------------------------------------------------------------------
__global__ __launch_bounds__(256) void transpose_x_kernel(
    const float* __restrict__ x, _Float16* __restrict__ xh) {
  __shared__ _Float16 tile[128 * HSTR];
  const int bh = blockIdx.x;             // b*128 + h
  const int b = bh >> 7, h = bh & 127;
  const float* src = x + (size_t)b * 1048576 + (size_t)h * 128;  // + c*16384 + w
  for (int id = threadIdx.x; id < 8192; id += 256) {
    const int c = id >> 7, w = id & 127;                  // coalesced along w
    tile[w * HSTR + c] = (_Float16)src[(size_t)c * 16384 + w];
  }
  __syncthreads();
  _Float16* dst = xh + (size_t)bh * 8192;                  // [w][c] contiguous
  for (int id = threadIdx.x; id < 1024; id += 256) {
    const int w = id >> 3, ch = id & 7;
    *(v8h*)(dst + w * 64 + ch * 8) = *(const v8h*)(tile + w * HSTR + ch * 8);
  }
}

// ---------------------------------------------------------------------------
// Kernel 2: parallel input projection  pre = in . Wih^T + bias.
// Sequence-tile layout shared with the RNN kernels: tile = b*8 + subtile,
// rows m = 16 sequences of the tile, element (m,t,k) at
//   in + (tile>>3)*1048576 + (tile&7)*16*mStride + m*mStride + t*tStride + k
// stage X: mStride=8192, tStride=64 ; stage Y: mStride=64, tStride=8192.
// Output in C-fragment layout: pre[((tile*128 + t)*64 + n)*16 + m], f32,
// so the RNN lane (column n, rows 8*ksel..) reads one aligned v8f.
// Grid: 2048 blocks x 128 threads; each block does 8 time steps.
// ---------------------------------------------------------------------------
__global__ __launch_bounds__(128) void pregemm_kernel(
    const _Float16* __restrict__ in, const _Float16* __restrict__ Wih,
    const float* __restrict__ bias, float* __restrict__ pre, int mStride,
    int tStride) {
  const int tid = threadIdx.x;
  const int lane = tid & 31;
  const int wave = tid >> 5;
  const int mrow = lane & 15;
  const int ksel = lane >> 4;
  const int n = wave * 16 + mrow;

  const int tile = blockIdx.x >> 4;
  const int t0 = (blockIdx.x & 15) << 3;

  const _Float16* base = in + (size_t)(tile >> 3) * 1048576 +
                         (size_t)(tile & 7) * 16 * (size_t)mStride +
                         (size_t)mrow * (size_t)mStride;

  // B fragments: lane L holds column n, K = kbase + ksel*16 + 0..15 (row of W).
  const v16h b0 = *(const v16h*)(Wih + n * 64 + ksel * 16);
  const v16h b1 = *(const v16h*)(Wih + n * 64 + 32 + ksel * 16);
  const float bv = bias[n];

  float* outp = pre + (size_t)tile * 131072 + (size_t)n * 16 + (ksel << 3);

  for (int tt = 0; tt < 8; ++tt) {
    const int t = t0 + tt;
    const _Float16* rp = base + (size_t)t * (size_t)tStride;
    const v16h a0 = combine(*(const v8h*)(rp + ksel * 16),
                            *(const v8h*)(rp + 8 + ksel * 16));
    const v16h a1 = combine(*(const v8h*)(rp + 32 + ksel * 16),
                            *(const v8h*)(rp + 40 + ksel * 16));
    v8f acc;
    for (int i = 0; i < 8; ++i) acc[i] = bv;
    acc = wmma_acc(a0, b0, acc);
    acc = wmma_acc(a1, b1, acc);
    *(v8f*)(outp + (size_t)t * 1024) = acc;
  }
}

// ---------------------------------------------------------------------------
// Kernel 3: X-axis serial recurrence. 128 blocks x 128 threads (4 waves).
// Block = 16 row-sequences; wave w owns hidden cols 16w..16w+15.
// h_t = tanh(pre_t + h_{t-1} . Whh^T); h streamed to hx (f16 [B,H,W,HX]).
// ---------------------------------------------------------------------------
__global__ __launch_bounds__(128) void rnn_x_kernel(
    const float* __restrict__ pre, const _Float16* __restrict__ Whh,
    _Float16* __restrict__ outx) {
  __shared__ _Float16 hbuf[16 * HSTR];
  const int tid = threadIdx.x;
  const int lane = tid & 31;
  const int wave = tid >> 5;
  const int mrow = lane & 15;
  const int ksel = lane >> 4;
  const int n = wave * 16 + mrow;

  _Float16* outb = outx + ((size_t)(blockIdx.x >> 3) * 128 +
                           (size_t)((blockIdx.x & 7) << 4)) * 8192;
  const float* preb =
      pre + (size_t)blockIdx.x * 131072 + (size_t)n * 16 + (ksel << 3);

  const v16h bhh0 = *(const v16h*)(Whh + n * 64 + ksel * 16);
  const v16h bhh1 = *(const v16h*)(Whh + n * 64 + 32 + ksel * 16);

  const int lm = tid >> 3;  // row for cooperative stores
  const int lj = tid & 7;   // 16B chunk

  for (int i = tid; i < 16 * HSTR; i += 128) hbuf[i] = (_Float16)0.f;

  const v8f vzero = {};
  v8f pcur = *(const v8f*)preb;
  v8f pnext = pcur;

#pragma unroll 1
  for (int t = 0; t < 128; ++t) {
    if (t + 1 < 128) pnext = *(const v8f*)(preb + (size_t)(t + 1) * 1024);
    __syncthreads();  // h_{t-1} writes visible

    if (t > 0) {  // stream h_{t-1} tile to global (contiguous 16B/thread)
      *(v8h*)(outb + (size_t)lm * 8192 + (size_t)(t - 1) * 64 + lj * 8) =
          *(const v8h*)(hbuf + lm * HSTR + lj * 8);
    }

    const v16h ah0 = load_a_lds(hbuf, mrow, ksel, 0);
    const v16h ah1 = load_a_lds(hbuf, mrow, ksel, 32);
    const v8f acc0 = wmma_acc(ah0, bhh0, pcur);   // independent accumulators:
    const v8f acc1 = wmma_acc(ah1, bhh1, vzero);  // the two WMMAs pipeline

    _Float16 hv[8];
    for (int i = 0; i < 8; ++i) hv[i] = (_Float16)dev_tanh(acc0[i] + acc1[i]);

    __syncthreads();  // all hbuf reads done before overwrite
    const int mb = ksel << 3;  // D rows: VGPR r -> M = r + 8*ksel
    for (int i = 0; i < 8; ++i) hbuf[(mb + i) * HSTR + n] = hv[i];
    pcur = pnext;
  }
  __syncthreads();
  *(v8h*)(outb + (size_t)lm * 8192 + (size_t)127 * 64 + lj * 8) =
      *(const v8h*)(hbuf + lm * HSTR + lj * 8);
}

// ---------------------------------------------------------------------------
// Kernel 4: Y-axis serial recurrence. Block = 16 column-sequences (b,w0..+15).
// Output written directly from D fragments: lane holds (hy, 8 consecutive w)
// -> one aligned 32B f32 store into out[B,HY,H,W].
// ---------------------------------------------------------------------------
__global__ __launch_bounds__(128) void rnn_y_kernel(
    const float* __restrict__ pre, const _Float16* __restrict__ Whh,
    float* __restrict__ out) {
  __shared__ _Float16 hbuf[16 * HSTR];
  const int tid = threadIdx.x;
  const int lane = tid & 31;
  const int wave = tid >> 5;
  const int mrow = lane & 15;
  const int ksel = lane >> 4;
  const int n = wave * 16 + mrow;

  const int b  = blockIdx.x >> 3;
  const int w0 = (blockIdx.x & 7) << 4;
  const float* preb =
      pre + (size_t)blockIdx.x * 131072 + (size_t)n * 16 + (ksel << 3);
  // out[((b*64 + n)*128 + t)*128 + w0 + 8*ksel + r]
  float* outp = out + (size_t)b * 1048576 + (size_t)n * 16384 + w0 + (ksel << 3);

  const v16h bhh0 = *(const v16h*)(Whh + n * 64 + ksel * 16);
  const v16h bhh1 = *(const v16h*)(Whh + n * 64 + 32 + ksel * 16);

  for (int i = tid; i < 16 * HSTR; i += 128) hbuf[i] = (_Float16)0.f;

  const v8f vzero = {};
  v8f pcur = *(const v8f*)preb;
  v8f pnext = pcur;

#pragma unroll 1
  for (int t = 0; t < 128; ++t) {
    if (t + 1 < 128) pnext = *(const v8f*)(preb + (size_t)(t + 1) * 1024);
    __syncthreads();

    const v16h ah0 = load_a_lds(hbuf, mrow, ksel, 0);
    const v16h ah1 = load_a_lds(hbuf, mrow, ksel, 32);
    const v8f acc0 = wmma_acc(ah0, bhh0, pcur);
    const v8f acc1 = wmma_acc(ah1, bhh1, vzero);

    v8f hf;
    _Float16 hv[8];
    for (int i = 0; i < 8; ++i) {
      const float th = dev_tanh(acc0[i] + acc1[i]);
      hf[i] = th;
      hv[i] = (_Float16)th;
    }
    *(v8f*)(outp + (size_t)t * 128) = hf;  // final f32 output, 32B/lane

    __syncthreads();
    const int mb = ksel << 3;
    for (int i = 0; i < 8; ++i) hbuf[(mb + i) * HSTR + n] = hv[i];
    pcur = pnext;
  }
}

// ---------------------------------------------------------------------------
// Launch. Workspace layout (needs 65536 + 2*33554432 + 67108864 = 134,283,264 B):
//   [0      , 32768 ) f16 weights (4 x 64x64)
//   [32768  , 33280 ) folded f32 biases (2 x 64)
//   [65536  , +32MB ) xh  : x transposed to [B,H,W,C] f16
//   [+32MB  , +64MB ) hx  : X-RNN hidden states [B,H,W,HX] f16
//   [+64MB  , +128MB) pre : f32 C-fragment-layout projections (reused X then Y)
// ---------------------------------------------------------------------------
extern "C" void kernel_launch(void* const* d_in, const int* in_sizes, int n_in,
                              void* d_out, int out_size, void* d_ws,
                              size_t ws_size, hipStream_t stream) {
  (void)in_sizes; (void)n_in; (void)out_size; (void)ws_size;
  const float* x     = (const float*)d_in[0];
  const float* Wih_x = (const float*)d_in[1];
  const float* Whh_x = (const float*)d_in[2];
  const float* bih_x = (const float*)d_in[3];
  const float* bhh_x = (const float*)d_in[4];
  const float* Wih_y = (const float*)d_in[5];
  const float* Whh_y = (const float*)d_in[6];
  const float* bih_y = (const float*)d_in[7];
  const float* bhh_y = (const float*)d_in[8];
  float* out = (float*)d_out;

  char* ws = (char*)d_ws;
  _Float16* wbuf = (_Float16*)ws;
  float*    bbuf = (float*)(ws + 32768);
  _Float16* xh   = (_Float16*)(ws + 65536);
  _Float16* hx   = (_Float16*)(ws + 65536 + 33554432);
  float*    pre  = (float*)(ws + 65536 + 2 * 33554432);

  prep_weights_kernel<<<1, 256, 0, stream>>>(Wih_x, Whh_x, bih_x, bhh_x, Wih_y,
                                             Whh_y, bih_y, bhh_y, wbuf, bbuf);
  transpose_x_kernel<<<2048, 256, 0, stream>>>(x, xh);

  // Stage X: wide projection GEMM, then the serial recurrence.
  pregemm_kernel<<<2048, 128, 0, stream>>>(xh, wbuf, bbuf, pre,
                                           /*mStride=*/8192, /*tStride=*/64);
  rnn_x_kernel<<<128, 128, 0, stream>>>(pre, wbuf + 4096, hx);

  // Stage Y: projection of the X-RNN output, then the serial recurrence.
  pregemm_kernel<<<2048, 128, 0, stream>>>(hx, wbuf + 8192, bbuf + 64, pre,
                                           /*mStride=*/64, /*tStride=*/8192);
  rnn_y_kernel<<<128, 128, 0, stream>>>(pre, wbuf + 12288, out);
}
